// DecoderBlock_50525995270856
// MI455X (gfx1250) — matrix operand order repaired and checked by reading, not compiled, not run on hardware
//
#include <hip/hip_runtime.h>
#include <hip/hip_bf16.h>
#include <stdint.h>

// Problem dims (fixed by reference)
#define BB   4
#define SS   2048
#define EE   1024
#define HH   16
#define DD   64
#define NTOK (BB * SS)   // 8192 tokens
#define FF   (4 * EE)    // 4096
#define QKVW (3 * EE)    // 3072

typedef __bf16 bf16;
typedef __attribute__((ext_vector_type(16))) __bf16 bf16x16;
typedef __attribute__((ext_vector_type(8)))  __bf16 bf16x8;
typedef __attribute__((ext_vector_type(8)))  float  f32x8;
typedef __attribute__((ext_vector_type(4)))  float  f32x4;

// ---------------------------------------------------------------------------
// CDNA5 async global->LDS copy (ASYNCcnt-tracked) + waits, via inline asm.
// ---------------------------------------------------------------------------
__device__ __forceinline__ uint32_t lds_addr32(const void* p) {
  // Low 32 bits of a generic pointer into LDS aperture == LDS byte offset.
  return (uint32_t)(uintptr_t)p;
}
__device__ __forceinline__ void async_b128(uint32_t lds, const bf16* g) {
  asm volatile("global_load_async_to_lds_b128 %0, %1, off"
               :: "v"(lds), "v"(g) : "memory");
}
__device__ __forceinline__ void wait_async0() {
  asm volatile("s_wait_asynccnt 0" ::: "memory");
}
// Wait until at most the 8 most recent async LDS loads are still in flight
// (ASYNCcnt completes in order -> the previous stage has fully landed).
__device__ __forceinline__ void wait_async8() {
  asm volatile("s_wait_asynccnt 8" ::: "memory");
}

__device__ __forceinline__ f32x8 wmma_bf16(bf16x16 a, bf16x16 b, f32x8 c) {
  return __builtin_amdgcn_wmma_f32_16x16x32_bf16(false, a, false, b,
                                                 (short)0, c, false, false);
}
__device__ __forceinline__ bf16x16 cat8(bf16x8 lo, bf16x8 hi) {
  return __builtin_shufflevector(lo, hi, 0,1,2,3,4,5,6,7,8,9,10,11,12,13,14,15);
}
__device__ __forceinline__ f32x8 zero8() {
  f32x8 z = {0.f,0.f,0.f,0.f,0.f,0.f,0.f,0.f};
  return z;
}

// ---------------------------------------------------------------------------
// Conversions / weight packing (vectorized: 8 elements per thread)
// ---------------------------------------------------------------------------
__global__ void k_f32_to_bf16(const float* __restrict__ in,
                              bf16* __restrict__ out, int n) {
  int i = (blockIdx.x * 256 + threadIdx.x) * 8;
  if (i >= n) return;
  f32x4 a = *(const f32x4*)(in + i);
  f32x4 b = *(const f32x4*)(in + i + 4);
  bf16x8 o;
#pragma unroll
  for (int j = 0; j < 4; ++j) { o[j] = (bf16)a[j]; o[4 + j] = (bf16)b[j]; }
  *(bf16x8*)(out + i) = o;
}

// Wq/Wk/Wv: [H,E,D] -> fused bf16 [E, 3E]; col h*D+d (q), +E (k), +2E (v)
__global__ void k_pack_qkv(const float* __restrict__ Wq,
                           const float* __restrict__ Wk,
                           const float* __restrict__ Wv,
                           bf16* __restrict__ Wf) {
  int idx = blockIdx.x * 256 + threadIdx.x;   // < H*E*(D/8) = 131072
  int d0 = (idx & 7) * 8;
  int e  = (idx >> 3) & (EE - 1);
  int h  = idx >> 13;
  size_t src = ((size_t)(h * EE + e)) * DD + d0;
  size_t dst = (size_t)e * QKVW + h * DD + d0;
  const float* s[3] = {Wq + src, Wk + src, Wv + src};
  bf16* o[3] = {Wf + dst, Wf + dst + EE, Wf + dst + 2 * EE};
#pragma unroll
  for (int m = 0; m < 3; ++m) {
    f32x4 a = *(const f32x4*)(s[m]);
    f32x4 b = *(const f32x4*)(s[m] + 4);
    bf16x8 v;
#pragma unroll
    for (int j = 0; j < 4; ++j) { v[j] = (bf16)a[j]; v[4 + j] = (bf16)b[j]; }
    *(bf16x8*)o[m] = v;
  }
}

// ---------------------------------------------------------------------------
// WMMA GEMM: C[M,N] = A[M,K]_bf16 @ B[K,N]_bf16  (+bias +residual +relu)
// Block tile 128x128, K-stage 64, double-buffered async LDS pipeline.
// 256 threads = 8 wave32 (4x2 wave grid), each wave computes 32x64.
// ---------------------------------------------------------------------------
template <bool HAS_BIAS, bool HAS_RES, bool RELU, bool OUTB, bool OUTF>
__global__ __launch_bounds__(256)
void k_gemm(const bf16* __restrict__ A, const bf16* __restrict__ Bm,
            int M, int N, int K,
            const float* __restrict__ bias,
            const float* __restrict__ resid, int ldr,
            bf16* __restrict__ outb, float* __restrict__ outf, int ldc) {
  __shared__ __attribute__((aligned(16))) bf16 sA[2][128][64];   // 2 x 16 KB
  __shared__ __attribute__((aligned(16))) bf16 sB[2][64][128];   // 2 x 16 KB

  const int t    = threadIdx.x;
  const int lane = t & 31;
  const int wid  = t >> 5;
  const int hf   = lane >> 4;    // wave half (K/row-half selector)
  const int ln   = lane & 15;
  const int wm   = wid & 3;      // 0..3
  const int wn   = wid >> 2;     // 0..1
  const int m0   = blockIdx.y * 128;
  const int n0   = blockIdx.x * 128;

  f32x8 acc[2][4];
#pragma unroll
  for (int mt = 0; mt < 2; ++mt)
#pragma unroll
    for (int nt = 0; nt < 4; ++nt) acc[mt][nt] = zero8();

  // Issue one 64-wide K stage (8 async b128 per thread: 4 for A, 4 for B)
  auto issue = [&](int k0, int buf) {
#pragma unroll
    for (int i = 0; i < 4; ++i) {
      int c  = t + 256 * i;                  // 0..1023
      int ra = c >> 3, ca = (c & 7) * 8;     // A: 128 rows x 8 chunks
      async_b128(lds_addr32(&sA[buf][ra][ca]),
                 A + (size_t)(m0 + ra) * K + k0 + ca);
      int rb = c >> 4, cb = (c & 15) * 8;    // B: 64 rows x 16 chunks
      async_b128(lds_addr32(&sB[buf][rb][cb]),
                 Bm + (size_t)(k0 + rb) * N + n0 + cb);
    }
  };

  const int nk = K >> 6;
  issue(0, 0);
  for (int ks = 0; ks < nk; ++ks) {
    const int buf = ks & 1;
    if (ks + 1 < nk) { issue((ks + 1) << 6, buf ^ 1); wait_async8(); }
    else             { wait_async0(); }
    __syncthreads();

#pragma unroll
    for (int kk = 0; kk < 2; ++kk) {
      bf16x16 af[2];
#pragma unroll
      for (int mt = 0; mt < 2; ++mt) {
        const bf16* row = &sA[buf][wm * 32 + mt * 16 + ln][kk * 32];
        bf16x8 lo = *(const bf16x8*)(row + hf * 8);
        bf16x8 hi = *(const bf16x8*)(row + 16 + hf * 8);
        af[mt] = cat8(lo, hi);
      }
#pragma unroll
      for (int nt = 0; nt < 4; ++nt) {
        bf16x16 bf;
#pragma unroll
        for (int s = 0; s < 16; ++s)
          bf[s] = sB[buf][kk * 32 + hf * 8 + (s & 7) + (s >> 3) * 16][wn * 64 + nt * 16 + ln];
#pragma unroll
        for (int mt = 0; mt < 2; ++mt)
          acc[mt][nt] = wmma_bf16(af[mt], bf, acc[mt][nt]);
      }
    }
    __syncthreads();
  }

  // Epilogue (C layout: VGPR j -> row j + 8*hf, lane%16 -> col)
#pragma unroll
  for (int nt = 0; nt < 4; ++nt) {
    const int col = n0 + wn * 64 + nt * 16 + ln;
    float bv = 0.f;
    if constexpr (HAS_BIAS) bv = bias[col];
#pragma unroll
    for (int mt = 0; mt < 2; ++mt)
#pragma unroll
      for (int j = 0; j < 8; ++j) {
        const int row = m0 + wm * 32 + mt * 16 + hf * 8 + j;
        float v = acc[mt][nt][j];
        if constexpr (HAS_BIAS) v += bv;
        if constexpr (HAS_RES)  v += resid[(size_t)row * ldr + col];
        if constexpr (RELU)     v = v > 0.f ? v : 0.f;
        if constexpr (OUTF)     outf[(size_t)row * ldc + col] = v;
        if constexpr (OUTB)     outb[(size_t)row * ldc + col] = (bf16)v;
      }
  }
}

// ---------------------------------------------------------------------------
// Flash attention (causal). One block per (b, h, 128-query tile); 8 waves,
// each owns 16 query rows. K/V double-buffered in 32-token tiles: next K is
// prefetched to registers and next V issued async-to-LDS before computing the
// current block. Diagonal-only masking; scale folded into the exp argument.
// ---------------------------------------------------------------------------
__global__ __launch_bounds__(256)
void k_attn(const bf16* __restrict__ qkv, bf16* __restrict__ attnb) {
  __shared__ __attribute__((aligned(16))) bf16 sKt[2][64][32];  // K^T [d][tok]
  __shared__ __attribute__((aligned(16))) bf16 sV[2][32][64];   // V  [tok][d]
  __shared__ __attribute__((aligned(16))) bf16 sP[8][16][32];   // per-wave P

  const int t    = threadIdx.x;
  const int lane = t & 31;
  const int wid  = t >> 5;
  const int hf   = lane >> 4;
  const int ln   = lane & 15;
  const int qb   = blockIdx.x;           // 0..15
  const int h    = blockIdx.y;           // 0..15
  const int b    = blockIdx.z;           // 0..3
  const size_t rowbase = (size_t)b * SS;
  const int s0      = qb * 128;
  const int myrow_s = s0 + wid * 16;     // wave's first query position
  const float scale = 0.03125f;          // 1024^-0.5

  // Cooperative K/V staging coordinates (per thread)
  const int tok  = t >> 3;               // 0..31
  const int doff = (t & 7) * 8;          // 0..56
  const bf16* kvbase = qkv + (rowbase + tok) * QKVW + h * DD + doff;

  // Q fragments for this wave's 16 rows (A-matrix, two 32-wide K steps)
  bf16x16 qf[2];
  {
    const bf16* qrow = qkv + (rowbase + myrow_s + ln) * QKVW + h * DD;
#pragma unroll
    for (int kk = 0; kk < 2; ++kk) {
      bf16x8 lo = *(const bf16x8*)(qrow + kk * 32 + hf * 8);
      bf16x8 hi = *(const bf16x8*)(qrow + kk * 32 + 16 + hf * 8);
      qf[kk] = cat8(lo, hi);
    }
  }

  f32x8 o[4];
#pragma unroll
  for (int nt = 0; nt < 4; ++nt) o[nt] = zero8();
  float mrow[8], lrow[8];
#pragma unroll
  for (int j = 0; j < 8; ++j) { mrow[j] = -__builtin_inff(); lrow[j] = 0.f; }

  const int nblk = (s0 + 128) >> 5;

  // Prologue: stage block 0 (K via register scatter, V via async DMA)
  bf16x8 kreg = *(const bf16x8*)(kvbase + EE);
  async_b128(lds_addr32(&sV[0][tok][doff]), kvbase + 2 * EE);
#pragma unroll
  for (int i2 = 0; i2 < 8; ++i2) sKt[0][doff + i2][tok] = kreg[i2];
  wait_async0();
  __syncthreads();

  for (int i = 0; i < nblk; ++i) {
    const int kb   = i << 5;
    const int buf  = i & 1;
    const bool more = (i + 1) < nblk;

    // Prefetch next block: K -> registers (in flight), V -> async LDS
    if (more) {
      kreg = *(const bf16x8*)(kvbase + (size_t)(kb + 32) * QKVW + EE);
      async_b128(lds_addr32(&sV[buf ^ 1][tok][doff]),
                 kvbase + (size_t)(kb + 32) * QKVW + 2 * EE);
    }

    // Wave-uniform skip of fully-masked blocks (barriers stay uniform)
    if (kb <= myrow_s + 15) {
      // Scores: q(16x64) @ k^T(64x32) -> 16x32 f32 (two 16x16 tiles), raw
      f32x8 sc[2];
#pragma unroll
      for (int nt = 0; nt < 2; ++nt) {
        sc[nt] = zero8();
#pragma unroll
        for (int kk = 0; kk < 2; ++kk) {
          bf16x16 kf;
#pragma unroll
          for (int s = 0; s < 16; ++s)
            kf[s] = sKt[buf][kk * 32 + hf * 8 + (s & 7) + (s >> 3) * 16][nt * 16 + ln];
          sc[nt] = wmma_bf16(qf[kk], kf, sc[nt]);
        }
      }

      // Causal mask only on diagonal blocks (wave-uniform branch)
      if (kb + 31 > myrow_s) {
#pragma unroll
        for (int j = 0; j < 8; ++j) {
          int qs = myrow_s + hf * 8 + j;
#pragma unroll
          for (int nt = 0; nt < 2; ++nt) {
            int ks = kb + nt * 16 + ln;
            sc[nt][j] = (ks <= qs) ? sc[nt][j] : -__builtin_inff();
          }
        }
      }

      // Row max on raw scores (reduce across the 16-lane half-group)
      float rmax[8];
#pragma unroll
      for (int j = 0; j < 8; ++j) {
        float m = fmaxf(sc[0][j], sc[1][j]);
        m = fmaxf(m, __shfl_xor(m, 1, 32));
        m = fmaxf(m, __shfl_xor(m, 2, 32));
        m = fmaxf(m, __shfl_xor(m, 4, 32));
        m = fmaxf(m, __shfl_xor(m, 8, 32));
        rmax[j] = m;
      }

      // Online softmax (running max kept in the scaled domain; one fma+exp
      // per element: p = exp(s*scale - mnew))
#pragma unroll
      for (int j = 0; j < 8; ++j) {
        float mnew = fmaxf(mrow[j], scale * rmax[j]);
        float f = __expf(mrow[j] - mnew);   // 0 on first block (mrow=-inf)
#pragma unroll
        for (int nt = 0; nt < 4; ++nt) o[nt][j] *= f;
        float rsum = 0.f;
#pragma unroll
        for (int nt = 0; nt < 2; ++nt) {
          float p = __expf(__builtin_fmaf(sc[nt][j], scale, -mnew));
          sc[nt][j] = p;
          rsum += p;
        }
        rsum += __shfl_xor(rsum, 1, 32);
        rsum += __shfl_xor(rsum, 2, 32);
        rsum += __shfl_xor(rsum, 4, 32);
        rsum += __shfl_xor(rsum, 8, 32);
        lrow[j] = lrow[j] * f + rsum;
        mrow[j] = mnew;
      }

      // P -> LDS (C layout -> A layout transpose), then O += P @ V
#pragma unroll
      for (int j = 0; j < 8; ++j)
#pragma unroll
        for (int nt = 0; nt < 2; ++nt)
          sP[wid][hf * 8 + j][nt * 16 + ln] = (bf16)sc[nt][j];

      bf16x16 pf;
      {
        const bf16* pr = &sP[wid][ln][0];
        bf16x8 lo = *(const bf16x8*)(pr + hf * 8);
        bf16x8 hi = *(const bf16x8*)(pr + 16 + hf * 8);
        pf = cat8(lo, hi);
      }
#pragma unroll
      for (int nt = 0; nt < 4; ++nt) {
        bf16x16 vf;
#pragma unroll
        for (int s = 0; s < 16; ++s)
          vf[s] = sV[buf][hf * 8 + (s & 7) + (s >> 3) * 16][nt * 16 + ln];
        o[nt] = wmma_bf16(pf, vf, o[nt]);
      }
    }

    // Commit next block's K scatter + make sure its async V landed
    if (more) {
#pragma unroll
      for (int i2 = 0; i2 < 8; ++i2) sKt[buf ^ 1][doff + i2][tok] = kreg[i2];
      wait_async0();
    }
    __syncthreads();
  }

  // Normalize and write concat-head output [NTOK, E] (bf16)
#pragma unroll
  for (int j = 0; j < 8; ++j) {
    float inv = 1.f / lrow[j];
    size_t rowg = rowbase + (size_t)(myrow_s + hf * 8 + j);
#pragma unroll
    for (int nt = 0; nt < 4; ++nt)
      attnb[rowg * EE + h * DD + nt * 16 + ln] = (bf16)(o[nt][j] * inv);
  }
}

// ---------------------------------------------------------------------------
// Host-side orchestration
// ---------------------------------------------------------------------------
static inline size_t align256(size_t x) { return (x + 255) & ~(size_t)255; }

extern "C" void kernel_launch(void* const* d_in, const int* in_sizes, int n_in,
                              void* d_out, int out_size, void* d_ws,
                              size_t ws_size, hipStream_t stream) {
  (void)in_sizes; (void)n_in; (void)out_size; (void)ws_size;
  const float* x  = (const float*)d_in[0];
  const float* Wq = (const float*)d_in[1];
  const float* Wk = (const float*)d_in[2];
  const float* Wv = (const float*)d_in[3];
  const float* Wo = (const float*)d_in[4];
  const float* bo = (const float*)d_in[5];
  const float* W1 = (const float*)d_in[6];
  const float* b1 = (const float*)d_in[7];
  const float* W2 = (const float*)d_in[8];
  const float* b2 = (const float*)d_in[9];
  float* out = (float*)d_out;

  // Workspace carve-up
  char* p = (char*)d_ws;
  bf16* xb    = (bf16*)p; p += align256((size_t)NTOK * EE * 2);
  bf16* wqkvb = (bf16*)p; p += align256((size_t)EE * QKVW * 2);
  bf16* wob   = (bf16*)p; p += align256((size_t)EE * EE * 2);
  bf16* w1b   = (bf16*)p; p += align256((size_t)EE * FF * 2);
  bf16* w2b   = (bf16*)p; p += align256((size_t)FF * EE * 2);
  bf16* qkvb  = (bf16*)p; p += align256((size_t)NTOK * QKVW * 2);
  bf16* attnb = (bf16*)p; p += align256((size_t)NTOK * EE * 2);
  float* y32  = (float*)p; p += align256((size_t)NTOK * EE * 4);
  bf16* yb    = (bf16*)p; p += align256((size_t)NTOK * EE * 2);
  bf16* hb    = (bf16*)p; p += align256((size_t)NTOK * FF * 2);

  // 1) bf16 conversions + QKV weight fusion (8 elems/thread)
  k_f32_to_bf16<<<(NTOK * EE / 8 + 255) / 256, 256, 0, stream>>>(x, xb, NTOK * EE);
  k_pack_qkv<<<(HH * EE * (DD / 8) + 255) / 256, 256, 0, stream>>>(Wq, Wk, Wv, wqkvb);
  k_f32_to_bf16<<<(EE * EE / 8 + 255) / 256, 256, 0, stream>>>(Wo, wob, EE * EE);
  k_f32_to_bf16<<<(EE * FF / 8 + 255) / 256, 256, 0, stream>>>(W1, w1b, EE * FF);
  k_f32_to_bf16<<<(FF * EE / 8 + 255) / 256, 256, 0, stream>>>(W2, w2b, FF * EE);

  // 2) Fused QKV projection: [8192,1024] @ [1024,3072] -> qkv bf16
  k_gemm<false, false, false, true, false>
      <<<dim3(QKVW / 128, NTOK / 128), 256, 0, stream>>>(
      xb, wqkvb, NTOK, QKVW, EE, nullptr, nullptr, 0, qkvb, nullptr, QKVW);

  // 3) Causal flash attention -> concat-head attn bf16 [8192,1024]
  k_attn<<<dim3(SS / 128, HH, BB), 256, 0, stream>>>(qkvb, attnb);

  // 4) Output projection + bias + residual(x) -> y (f32 + bf16)
  k_gemm<true, true, false, true, true>
      <<<dim3(EE / 128, NTOK / 128), 256, 0, stream>>>(
      attnb, wob, NTOK, EE, EE, bo, x, EE, yb, y32, EE);

  // 5) FFN1 + bias + ReLU -> h bf16 [8192,4096]
  k_gemm<true, false, true, true, false>
      <<<dim3(FF / 128, NTOK / 128), 256, 0, stream>>>(
      yb, w1b, NTOK, FF, EE, b1, nullptr, 0, hb, nullptr, FF);

  // 6) FFN2 + bias + residual(y) -> final f32 output
  k_gemm<true, true, false, false, true>
      <<<dim3(EE / 128, NTOK / 128), 256, 0, stream>>>(
      hb, w2b, NTOK, EE, FF, b2, y32, EE, nullptr, out, EE);
}
